// Hawk_21380347199572
// MI455X (gfx1250) — matrix operand order repaired and checked by reading, not compiled, not run on hardware
//
#include <hip/hip_runtime.h>
#include <hip/hip_bf16.h>
#include <cstdint>

// Hawk gated linear RNN for MI455X (gfx1250, wave32, WMMA, async-to-LDS).
// Pipeline: cvt->GEMM1(wmma bf16)->conv->GEMM2(wmma)->gate->scan(3 pass)->GEMM3(wmma)

constexpr int BATCH = 4;
constexpr int T     = 4096;
constexpr int D     = 1024;
constexpr int H     = 1536;
constexpr int H2    = 2 * H;       // 3072
constexpr int NT    = BATCH * T;   // 16384 rows
constexpr int NC    = 32;          // scan chunks per sequence
constexpr int CL    = T / NC;      // 128 steps per chunk

typedef __attribute__((ext_vector_type(16))) __bf16 v16bf;
typedef __attribute__((ext_vector_type(8)))  float  v8f;

union FragAB { uint4 u[2]; v16bf v; };

__device__ __forceinline__ unsigned short f2bf(float f) {
  unsigned u = __float_as_uint(f);
  unsigned r = 0x7FFFu + ((u >> 16) & 1u);
  return (unsigned short)((u + r) >> 16);
}

// CDNA5 async global->LDS copy (16B per lane), tracked with ASYNCcnt.
__device__ __forceinline__ void async_copy_b128(unsigned lds_off, const void* gptr) {
  asm volatile("global_load_async_to_lds_b128 %0, %1, off"
               :: "v"(lds_off), "v"((unsigned long long)(size_t)gptr)
               : "memory");
}
__device__ __forceinline__ void wait_async_zero() {
  asm volatile("s_wait_asynccnt 0x0" ::: "memory");
}

// ---------------- f32 -> bf16 bulk convert ----------------
__global__ void cvt_f32_bf16(const float* __restrict__ src,
                             unsigned short* __restrict__ dst, size_t n) {
  size_t i = (size_t)blockIdx.x * blockDim.x + threadIdx.x;
  size_t stride = (size_t)gridDim.x * blockDim.x;
  for (; i < n; i += stride) dst[i] = f2bf(src[i]);
}

// ---------------- generic bf16 WMMA GEMM: C(MxN) = A(MxK) * B(NxK)^T ----------------
// Block tile 128x128, BK=64, 256 threads = 8 waves (4x2), wave tile 32x64.
// Double-buffered LDS fed by global_load_async_to_lds_b128; 16 WMMAs per wave per tile.
// All 12 fragments of a tile live in distinct VGPRs so the ds_loads pipeline freely.
#define BM 128
#define BN 128
#define BK 64
#define LDSW (BK + 8)   // padded row stride (bf16 elems): 144B rows, 16B aligned

__global__ __launch_bounds__(256)
void wmma_gemm_bf16(const unsigned short* __restrict__ A, int lda,
                    const unsigned short* __restrict__ B, int ldb,
                    float* __restrict__ C, int ldc, int Kdim) {
  __shared__ unsigned short As[2][BM * LDSW];
  __shared__ unsigned short Bs[2][BN * LDSW];

  const int tid  = threadIdx.x;
  const int lane = tid & 31;
  const int wave = tid >> 5;      // 0..7
  const int wm   = wave >> 1;     // 0..3 -> M direction
  const int wn   = wave & 1;      // 0..1 -> N direction

  const int blockM = blockIdx.y * BM;
  const int blockN = blockIdx.x * BN;
  const int waveM  = wm * 32;
  const int waveN  = wn * 64;

  v8f acc[2][4];
  #pragma unroll
  for (int i = 0; i < 2; i++)
    #pragma unroll
    for (int j = 0; j < 4; j++) acc[i][j] = {};

  // staging map: tile = 128 rows x 8 groups of 8 bf16 (16B) = 1024 units, 4/thread
  const int urow = tid >> 3;   // 0..31
  const int ukg  = tid & 7;    // 0..7

  auto stage = [&](int buf, int k0) {
    #pragma unroll
    for (int u = 0; u < 4; u++) {
      const int row = urow + 32 * u;
      const int kel = ukg * 8;
      async_copy_b128((unsigned)(size_t)&As[buf][row * LDSW + kel],
                      A + (size_t)(blockM + row) * lda + k0 + kel);
      async_copy_b128((unsigned)(size_t)&Bs[buf][row * LDSW + kel],
                      B + (size_t)(blockN + row) * ldb + k0 + kel);
    }
  };

  // A-fragment lane layout (16-bit A 16x32): lanes 0-15: K0..7 + K16..23,
  // lanes 16-31: K8..15 + K24..31, row = lane&15.
  const int arow = lane & 15;
  const int aklo = (lane < 16) ? 0 : 8;
  // B-fragment (32x16): lane&15 = N, lanes<16 hold K0..7 + K16..23, lanes>=16 hold K8..15 + K24..31.
  const int bk   = (lane < 16) ? 0 : 8;

  const int ktiles = Kdim / BK;
  stage(0, 0);

  for (int kt = 0; kt < ktiles; kt++) {
    const int buf = kt & 1;
    wait_async_zero();        // this wave's async copies for tile kt are in LDS
    __syncthreads();          // everyone's copies are in LDS / prior reads done

    if (kt + 1 < ktiles) stage(buf ^ 1, (kt + 1) * BK);  // overlap with compute

    // Load the entire BK=64 tile's fragments into distinct registers up front
    // (2 substeps x (2 A-frags + 4 B-frags) = 96 VGPRs), then a burst of 16 WMMAs.
    FragAB af[2][2];
    FragAB bf[2][4];
    #pragma unroll
    for (int s = 0; s < 2; s++) {
      const int ks = s * 32;
      #pragma unroll
      for (int i = 0; i < 2; i++) {
        const unsigned short* p = &As[buf][(waveM + i * 16 + arow) * LDSW + ks];
        af[s][i].u[0] = *(const uint4*)(p + aklo);
        af[s][i].u[1] = *(const uint4*)(p + aklo + 16);
      }
      #pragma unroll
      for (int j = 0; j < 4; j++) {
        const unsigned short* p = &Bs[buf][(waveN + j * 16 + arow) * LDSW + ks + bk];
        bf[s][j].u[0] = *(const uint4*)(p);
        bf[s][j].u[1] = *(const uint4*)(p + 16);
      }
    }
    #pragma unroll
    for (int s = 0; s < 2; s++)
      #pragma unroll
      for (int j = 0; j < 4; j++)
        #pragma unroll
        for (int i = 0; i < 2; i++)
          acc[i][j] = __builtin_amdgcn_wmma_f32_16x16x32_bf16(
              false, af[s][i].v, false, bf[s][j].v, (short)0, acc[i][j], false, false);
    // no trailing barrier needed: next iteration's wait+barrier orders reuse
  }

  // C/D layout: VGPR r -> lanes 0-15: (M=r, N=lane), lanes 16-31: (M=r+8, N=lane-16)
  const int cn    = lane & 15;
  const int mhalf = (lane < 16) ? 0 : 8;
  #pragma unroll
  for (int i = 0; i < 2; i++) {
    #pragma unroll
    for (int j = 0; j < 4; j++) {
      float* cp = C + (size_t)(blockM + waveM + i * 16 + mhalf) * ldc
                    + (blockN + waveN + j * 16 + cn);
      #pragma unroll
      for (int r = 0; r < 8; r++) cp[(size_t)r * ldc] = acc[i][j][r];
    }
  }
}

// ---------------- depthwise causal conv (K=4) on xb = gx[:, H:2H] ----------------
__global__ void conv_kernel(const float* __restrict__ gx,
                            const float* __restrict__ conv_w,
                            const float* __restrict__ conv_b,
                            float* __restrict__ xbc,
                            unsigned short* __restrict__ xbc_bf) {
  size_t idx = (size_t)blockIdx.x * blockDim.x + threadIdx.x;
  if (idx >= (size_t)NT * H) return;
  int h   = (int)(idx % H);
  size_t row = idx / H;            // n*T + t
  int t   = (int)(row % T);
  float acc = conv_b[h];
  #pragma unroll
  for (int k = 0; k < 4; k++) {
    int tt = t - 3 + k;
    if (tt >= 0)
      acc += conv_w[h * 4 + k] * gx[(row - 3 + k) * H2 + H + h];
  }
  xbc[idx]    = acc;
  xbc_bf[idx] = f2bf(acc);
}

// ---------------- alpha / xs (in-place over fg buffer) ----------------
__global__ void gate_kernel(float* __restrict__ fg,
                            const float* __restrict__ xbc,
                            const float* __restrict__ b_g,
                            const float* __restrict__ fbase) {
  size_t idx = (size_t)blockIdx.x * blockDim.x + threadIdx.x;
  if (idx >= (size_t)NT * H) return;
  int h = (int)(idx % H);
  size_t row = idx / H;
  float f = fg[row * H2 + h]     + b_g[h];
  float i = fg[row * H2 + H + h] + b_g[H + h];
  float fb = fbase[h];
  float sp = (fb > 20.f) ? fb : log1pf(expf(fb));        // softplus
  float sf = 1.f / (1.f + expf(-f));                     // sigmoid(forget)
  float alpha = expf(-8.0f * sp * sf);
  float beta  = sqrtf(fmaxf(1.f - alpha * alpha + 1e-6f, 0.f));
  float si = 1.f / (1.f + expf(-i));
  fg[row * H2 + h]     = alpha;
  fg[row * H2 + H + h] = beta * si * xbc[idx];           // xs
}

// ---------------- chunked linear scan, pass 1: per-chunk carries ----------------
__global__ void scan_pass1(const float* __restrict__ fg,
                           float* __restrict__ Acar, float* __restrict__ Hcar) {
  size_t idx = (size_t)blockIdx.x * blockDim.x + threadIdx.x;
  if (idx >= (size_t)BATCH * H * NC) return;
  int h = (int)(idx % H);
  int c = (int)((idx / H) % NC);
  int n = (int)(idx / ((size_t)H * NC));
  size_t base = ((size_t)n * T + (size_t)c * CL) * H2 + h;
  float a = 1.f, hh = 0.f;
  for (int t = 0; t < CL; t++) {
    float al = fg[base];
    float xs = fg[base + H];
    a  *= al;
    hh  = al * hh + xs;
    base += H2;
  }
  size_t ci = ((size_t)n * H + h) * NC + c;
  Acar[ci] = a;
  Hcar[ci] = hh;
}

// ---------------- pass 2: propagate carries across chunks ----------------
__global__ void scan_pass2(const float* __restrict__ Acar,
                           const float* __restrict__ Hcar,
                           float* __restrict__ H0) {
  int idx = blockIdx.x * blockDim.x + threadIdx.x;
  if (idx >= BATCH * H) return;
  const float* a  = Acar + (size_t)idx * NC;
  const float* hc = Hcar + (size_t)idx * NC;
  float* h0 = H0 + (size_t)idx * NC;
  float h = 0.f;
  for (int c = 0; c < NC; c++) {
    h0[c] = h;
    h = a[c] * h + hc[c];
  }
}

// ---------------- pass 3: re-scan + gelu(gate)*h -> bf16 for GEMM3 ----------------
__global__ void scan_pass3(const float* __restrict__ fg,
                           const float* __restrict__ gx,
                           const float* __restrict__ H0,
                           unsigned short* __restrict__ ybf) {
  size_t idx = (size_t)blockIdx.x * blockDim.x + threadIdx.x;
  if (idx >= (size_t)BATCH * H * NC) return;
  int h = (int)(idx % H);
  int c = (int)((idx / H) % NC);
  int n = (int)(idx / ((size_t)H * NC));
  float hh = H0[((size_t)n * H + h) * NC + c];
  size_t base  = ((size_t)n * T + (size_t)c * CL) * H2 + h;   // alpha/xs in fg
  size_t gbase = base;                                        // gate col h in gx
  size_t ybase = ((size_t)n * T + (size_t)c * CL) * H + h;
  for (int t = 0; t < CL; t++) {
    float al = fg[base];
    float xs = fg[base + H];
    hh = al * hh + xs;
    float g = gx[gbase];
    float gelu = 0.5f * g * (1.f + erff(g * 0.70710678118f));  // exact gelu
    ybf[ybase] = f2bf(gelu * hh);
    base += H2; gbase += H2; ybase += H;
  }
}

extern "C" void kernel_launch(void* const* d_in, const int* in_sizes, int n_in,
                              void* d_out, int out_size, void* d_ws, size_t ws_size,
                              hipStream_t stream) {
  (void)in_sizes; (void)n_in; (void)out_size; (void)ws_size;
  const float* x      = (const float*)d_in[0];
  const float* W_in   = (const float*)d_in[1];
  const float* conv_w = (const float*)d_in[2];
  const float* conv_b = (const float*)d_in[3];
  const float* W_g    = (const float*)d_in[4];
  const float* b_g    = (const float*)d_in[5];
  const float* fbase  = (const float*)d_in[6];
  const float* W_out  = (const float*)d_in[7];
  float* out = (float*)d_out;

  char* w = (char*)d_ws;
  size_t off = 0;
  auto take = [&](size_t bytes) { char* p = w + off; off += (bytes + 255) & ~(size_t)255; return p; };

  unsigned short* xbf    = (unsigned short*)take((size_t)NT * D * 2);   //  32 MiB
  unsigned short* winbf  = (unsigned short*)take((size_t)H2 * D * 2);   //   6 MiB
  unsigned short* wgbf   = (unsigned short*)take((size_t)H2 * H * 2);   //   9 MiB
  unsigned short* woutbf = (unsigned short*)take((size_t)D * H * 2);    //   3 MiB
  float*          gx     = (float*)take((size_t)NT * H2 * 4);           // 192 MiB (gate|xb)
  float*          fg     = (float*)take((size_t)NT * H2 * 4);           // 192 MiB (->alpha|xs)
  float*          xbc    = (float*)take((size_t)NT * H * 4);            //  96 MiB
  unsigned short* xbcbf  = (unsigned short*)take((size_t)NT * H * 2);   //  48 MiB
  unsigned short* ybf    = (unsigned short*)take((size_t)NT * H * 2);   //  48 MiB
  float*          Acar   = (float*)take((size_t)BATCH * H * NC * 4);
  float*          Hcar   = (float*)take((size_t)BATCH * H * NC * 4);
  float*          H0     = (float*)take((size_t)BATCH * H * NC * 4);

  // 1) bf16 conversions
  cvt_f32_bf16<<<4096, 256, 0, stream>>>(x,     xbf,    (size_t)NT * D);
  cvt_f32_bf16<<<2048, 256, 0, stream>>>(W_in,  winbf,  (size_t)H2 * D);
  cvt_f32_bf16<<<2048, 256, 0, stream>>>(W_g,   wgbf,   (size_t)H2 * H);
  cvt_f32_bf16<<<1024, 256, 0, stream>>>(W_out, woutbf, (size_t)D * H);

  // 2) GEMM1: gx = x @ W_in^T   (16384 x 1024 x 3072)
  dim3 g1(H2 / BN, NT / BM);
  wmma_gemm_bf16<<<g1, 256, 0, stream>>>(xbf, D, winbf, D, gx, H2, D);

  // 3) depthwise causal conv on xb half
  const int EW = (int)(((size_t)NT * H + 255) / 256);  // 98304 blocks
  conv_kernel<<<EW, 256, 0, stream>>>(gx, conv_w, conv_b, xbc, xbcbf);

  // 4) GEMM2: fg = xbc @ W_g^T  (16384 x 1536 x 3072)
  wmma_gemm_bf16<<<g1, 256, 0, stream>>>(xbcbf, H, wgbf, H, fg, H2, H);

  // 5) alpha / xs (in place over fg)
  gate_kernel<<<EW, 256, 0, stream>>>(fg, xbc, b_g, fbase);

  // 6) chunk-parallel linear scan + gelu-gate fusion
  scan_pass1<<<(BATCH * H * NC + 255) / 256, 256, 0, stream>>>(fg, Acar, Hcar);
  scan_pass2<<<(BATCH * H + 255) / 256, 256, 0, stream>>>(Acar, Hcar, H0);
  scan_pass3<<<(BATCH * H * NC + 255) / 256, 256, 0, stream>>>(fg, gx, H0, ybf);

  // 7) GEMM3: out = y @ W_out^T (16384 x 1536 x 1024)
  dim3 g3(D / BN, NT / BM);
  wmma_gemm_bf16<<<g3, 256, 0, stream>>>(ybf, H, woutbf, H, out, D, H);
}
